// SuperpointNeuralOperator_16441134809176
// MI455X (gfx1250) — matrix operand order repaired and checked by reading, not compiled, not run on hardware
//
#include <hip/hip_runtime.h>

#define NPTS 40000
#define KNBR 16
#define HDIM 64
#define XSTR 168    // halves per X row (16B-aligned stride, cols 0..159 used)
#define W1STR 168   // halves per W1T column
#define W2STR 80    // halves per W2T/WWT column & H1 row
#define WPB 8       // waves per block (wave32)
#define TPB 256
#define GRID_STEP 512

typedef __attribute__((ext_vector_type(16))) _Float16 v16h;
typedef __attribute__((ext_vector_type(8)))  _Float16 v8h;
typedef __attribute__((ext_vector_type(4)))  _Float16 v4h;
typedef __attribute__((ext_vector_type(2)))  _Float16 v2h;
typedef __attribute__((ext_vector_type(8)))  float    v8f;

// intra-wave LDS fence: LDS ops are in-order per wave; this stops the
// compiler from reordering across cross-lane LDS data exchange.
#define LDS_FENCE() asm volatile("s_wait_dscnt 0" ::: "memory")

__device__ __forceinline__ float gelu_f(float x) {
  return 0.5f * x * (1.0f + erff(x * 0.70710678118654752f));
}
__device__ __forceinline__ float sigmoid_f(float x) {
  return 1.0f / (1.0f + __expf(-x));
}

__device__ __forceinline__ v8f wmma16(v16h a, v16h b, v8f c) {
  return __builtin_amdgcn_wmma_f32_16x16x32_f16(false, a, false, b, (short)0, c,
                                                false, false);
}

// A fragment (16x32 f16): lane holds row M=lane&15; halves[0..15]=K base..base+15,
// base = kchunk*32 + (lane>=16 ? 16 : 0).
__device__ __forceinline__ v16h frag_a(const _Float16* rowbase, int kchunk, int lane) {
  const _Float16* p = rowbase + kchunk * 32 + ((lane >> 4) << 4);
  v8h lo = *(const v8h*)p;
  v8h hi = *(const v8h*)(p + 8);
  return __builtin_shufflevector(lo, hi, 0, 1, 2, 3, 4, 5, 6, 7, 8, 9, 10, 11,
                                 12, 13, 14, 15);
}

// B fragment (32x16 f16): lane holds col n=lane&15 (within tile nt);
// halves[t]=K kchunk*32 + (lane>=16 ? 16 : 0) + t, t=0..15 (contiguous).
// Weight matrix stored column-major in LDS: wt[col*stride + k].
__device__ __forceinline__ v16h frag_b(const _Float16* wt, int stride, int kchunk,
                                       int nt, int lane) {
  const _Float16* p = wt + (nt * 16 + (lane & 15)) * stride + kchunk * 32 +
                      ((lane >> 4) << 4);
  v8h lo = *(const v8h*)p;
  v8h hi = *(const v8h*)(p + 8);
  return __builtin_shufflevector(lo, hi, 0, 1, 2, 3, 4, 5, 6, 7, 8, 9, 10, 11,
                                 12, 13, 14, 15);
}

__global__ __launch_bounds__(TPB) void lift_kernel(
    const float* __restrict__ coords, const float* __restrict__ feat,
    const float* __restrict__ lw, const float* __restrict__ lb,
    float* __restrict__ v0) {
  int tid = blockIdx.x * TPB + threadIdx.x;
  if (tid >= NPTS * HDIM) return;
  int i = tid >> 6;
  int c = tid & 63;
  float acc = lb[c];
#pragma unroll
  for (int k = 0; k < 3; ++k) acc += coords[i * 3 + k] * lw[k * 64 + c];
#pragma unroll
  for (int k = 0; k < 6; ++k) acc += feat[i * 6 + k] * lw[(3 + k) * 64 + c];
  v0[tid] = acc;
}

__global__ __launch_bounds__(TPB) void tail_copy(
    const int* __restrict__ idxp, const float* __restrict__ vfin,
    float* __restrict__ out_idx, float* __restrict__ out_v) {
  int tid = blockIdx.x * TPB + threadIdx.x;
  if (tid < NPTS * KNBR) out_idx[tid] = (float)idxp[tid];
  if (tid < NPTS * HDIM) out_v[tid] = vfin[tid];
}

__global__ __launch_bounds__(TPB) void sno_step(
    const float* __restrict__ vin, float* __restrict__ vout,
    const float* __restrict__ coords, const int* __restrict__ idxp,
    const float* __restrict__ w1, const float* __restrict__ b1,
    const float* __restrict__ w2, const float* __restrict__ b2,
    const float* __restrict__ w3, const float* __restrict__ b3,
    const float* __restrict__ Ww, const float* __restrict__ lng,
    const float* __restrict__ lnb, int t, const float* __restrict__ pw1,
    const float* __restrict__ pb1, const float* __restrict__ pw2,
    const float* __restrict__ pb2, float* __restrict__ out_scores,
    float* __restrict__ out_wij, int finalPass, int nWavesTotal) {
  __shared__ __align__(16) _Float16 sW1T[64 * W1STR];
  __shared__ __align__(16) _Float16 sW2T[64 * W2STR];
  __shared__ __align__(16) _Float16 sWWT[64 * W2STR];
  __shared__ __align__(16) _Float16 sX[WPB][16 * XSTR];
  __shared__ __align__(16) _Float16 sH1[WPB][16 * W2STR];
  __shared__ __align__(16) _Float16 sVI[WPB][64];
  __shared__ __align__(16) float sVW[WPB][64];
  __shared__ float sGB[WPB][16];
  __shared__ float sPW1[64 * 32];

  const int lane = threadIdx.x & 31;
  const int wv = threadIdx.x >> 5;

  // ---- stage weights into LDS (f16, column-major, K-permuted for W1) ----
  // X column layout: [v_i: 0..63 | v_j: 64..127 | rel: 128..130 | zeros]
  // orig kernel_input row for k' : k'<128 -> k'+3 ; 128..130 -> k'-128
  for (int s = threadIdx.x; s < 64 * 160; s += TPB) {
    int c = s / 160;
    int kp = s - c * 160;
    float val = 0.f;
    if (kp < 131) {
      int orig = (kp < 128) ? (kp + 3) : (kp - 128);
      val = w1[orig * 64 + c];
    }
    sW1T[c * W1STR + kp] = (_Float16)val;
  }
  for (int s = threadIdx.x; s < 64 * 64; s += TPB) {
    int c = s & 63;
    int k = s >> 6;
    sW2T[c * W2STR + k] = (_Float16)w2[k * 64 + c];
    sWWT[c * W2STR + k] = (_Float16)Ww[k * 64 + c];
  }
  if (finalPass) {
    for (int s = threadIdx.x; s < 64 * 32; s += TPB) sPW1[s] = pw1[s];
  }
  // zero-pad X cols 131..159 once (never overwritten afterwards)
  for (int s = lane; s < 16 * 29; s += 32) {
    int r = s / 29;
    int cc = 131 + (s - r * 29);
    sX[wv][r * XSTR + cc] = (_Float16)0.f;
  }
  __syncthreads();

  const int cL = lane & 15;  // column-in-tile / edge id
  const int hf = lane >> 4;  // half of the wave

  float b1c[4], b2c[4], w3c[4];
#pragma unroll
  for (int nt = 0; nt < 4; ++nt) {
    b1c[nt] = b1[nt * 16 + cL];
    b2c[nt] = b2[nt * 16 + cL];
    w3c[nt] = w3[nt * 16 + cL];
  }
  const float b3s = b3[0];
  float pw2c = 0.f, pb1c = 0.f, pb2s = 0.f;
  if (finalPass) {
    pw2c = pw2[lane];
    pb1c = pb1[lane];
    pb2s = pb2[0];
  }
  float lg0 = 0.f, lg1 = 0.f, lb0v = 0.f, lb1v = 0.f;
  if (!finalPass) {
    lg0 = lng[t * 64 + 2 * lane];
    lg1 = lng[t * 64 + 2 * lane + 1];
    lb0v = lnb[t * 64 + 2 * lane];
    lb1v = lnb[t * 64 + 2 * lane + 1];
  }

  // ---- hoist loop-invariant B fragments (W2, W_w) into registers ----
  // 16 fragments x 8 VGPRs = 128 VGPRs resident; W1 (20 frags) stays in LDS
  // to remain inside the directly-addressable VGPR range.
  v16h bW2[2][4], bWW[2][4];
#pragma unroll
  for (int c = 0; c < 2; ++c)
#pragma unroll
    for (int nt = 0; nt < 4; ++nt) {
      bW2[c][nt] = frag_b(sW2T, W2STR, c, nt, lane);
      bWW[c][nt] = frag_b(sWWT, W2STR, c, nt, lane);
    }

  _Float16* Xw = &sX[wv][0];
  _Float16* H1w = &sH1[wv][0];
  _Float16* VIw = &sVI[wv][0];
  float* VWw = &sVW[wv][0];

  const int wgid = blockIdx.x * WPB + wv;

  for (int i = wgid; i < NPTS; i += nWavesTotal) {
    // ---- stage v_i (f16) ----
    float2 vv = *(const float2*)(vin + i * 64 + 2 * lane);
    {
      v2h p;
      p[0] = (_Float16)vv.x;
      p[1] = (_Float16)vv.y;
      *(v2h*)(VIw + 2 * lane) = p;
    }
    if (finalPass) {
      VWw[2 * lane] = vv.x;
      VWw[2 * lane + 1] = vv.y;
    }
    LDS_FENCE();

    // ---- stage X rows: lane handles edge e=cL, half hf ----
    const int e = cL;
    const int j = idxp[i * KNBR + e];
    {
      const _Float16* src = VIw + hf * 32;
      _Float16* dst = Xw + e * XSTR + hf * 32;
#pragma unroll
      for (int q = 0; q < 4; ++q) *(v8h*)(dst + 8 * q) = *(const v8h*)(src + 8 * q);
    }
    {
      const float4* vj4 = (const float4*)(vin + j * 64 + hf * 32);
      _Float16* dst = Xw + e * XSTR + 64 + hf * 32;
#pragma unroll
      for (int q = 0; q < 8; ++q) {
        float4 f = vj4[q];
        v4h p;
        p[0] = (_Float16)f.x;
        p[1] = (_Float16)f.y;
        p[2] = (_Float16)f.z;
        p[3] = (_Float16)f.w;
        *(v4h*)(dst + 4 * q) = p;
      }
    }
    if (hf == 0) {
      float rx = coords[j * 3 + 0] - coords[i * 3 + 0];
      float ry = coords[j * 3 + 1] - coords[i * 3 + 1];
      float rz = coords[j * 3 + 2] - coords[i * 3 + 2];
      Xw[e * XSTR + 128] = (_Float16)rx;
      Xw[e * XSTR + 129] = (_Float16)ry;
      Xw[e * XSTR + 130] = (_Float16)rz;
    }
    LDS_FENCE();

    // ---- MLP1: h1 = gelu(X @ W1 + b1), 16x160 @ 160x64 ----
    v8f acc[4] = {};
    const _Float16* arow = Xw + cL * XSTR;
#pragma unroll
    for (int c = 0; c < 5; ++c) {
      v16h a = frag_a(arow, c, lane);
#pragma unroll
      for (int nt = 0; nt < 4; ++nt)
        acc[nt] = wmma16(a, frag_b(sW1T, W1STR, c, nt, lane), acc[nt]);
    }
    const int m0 = hf * 8;
#pragma unroll
    for (int nt = 0; nt < 4; ++nt) {
      int col = nt * 16 + cL;
#pragma unroll
      for (int r = 0; r < 8; ++r) {
        float x = gelu_f(acc[nt][r] + b1c[nt]);
        H1w[(m0 + r) * W2STR + col] = (_Float16)x;
      }
    }
    LDS_FENCE();

    // ---- MLP2: h2pre = h1 @ W2, 16x64 @ 64x64 (B resident in VGPRs) ----
    v8f acc2[4] = {};
    const _Float16* hrow = H1w + cL * W2STR;
#pragma unroll
    for (int c = 0; c < 2; ++c) {
      v16h a = frag_a(hrow, c, lane);
#pragma unroll
      for (int nt = 0; nt < 4; ++nt)
        acc2[nt] = wmma16(a, bW2[c][nt], acc2[nt]);
    }

    // ---- g = sigmoid(gelu(h2pre+b2) @ w3 + b3) per edge ----
    float part[8];
#pragma unroll
    for (int r = 0; r < 8; ++r) part[r] = 0.f;
#pragma unroll
    for (int nt = 0; nt < 4; ++nt)
#pragma unroll
      for (int r = 0; r < 8; ++r)
        part[r] += gelu_f(acc2[nt][r] + b2c[nt]) * w3c[nt];
#pragma unroll
    for (int m = 1; m < 16; m <<= 1)
#pragma unroll
      for (int r = 0; r < 8; ++r) part[r] += __shfl_xor(part[r], m, 32);
    float g[8], go[8];
#pragma unroll
    for (int r = 0; r < 8; ++r) g[r] = sigmoid_f(part[r] + b3s);
#pragma unroll
    for (int r = 0; r < 8; ++r) go[r] = __shfl_xor(g[r], 16, 32);
    float gl[8], gh[8];
#pragma unroll
    for (int r = 0; r < 8; ++r) {
      gl[r] = hf ? go[r] : g[r];  // edges 0..7
      gh[r] = hf ? g[r] : go[r];  // edges 8..15
    }

    if (finalPass) {
      // ---- w_ij ----
      if (lane == 0) {
#pragma unroll
        for (int r = 0; r < 8; ++r) sGB[wv][r] = g[r];
      }
      if (lane == 16) {
#pragma unroll
        for (int r = 0; r < 8; ++r) sGB[wv][8 + r] = g[r];
      }
      LDS_FENCE();
      if (lane < 16) out_wij[i * KNBR + lane] = sGB[wv][lane];
      // ---- scores = sigmoid(gelu(v@pw1+pb1)@pw2+pb2) ----
      float s1 = pb1c;
      for (int k2 = 0; k2 < 64; ++k2) s1 += VWw[k2] * sPW1[k2 * 32 + lane];
      float sg = gelu_f(s1) * pw2c;
#pragma unroll
      for (int m = 1; m < 32; m <<= 1) sg += __shfl_xor(sg, m, 32);
      if (lane == 0) out_scores[i] = sigmoid_f(sg + pb2s);
    } else {
      // ---- integral = mean_k(g * v_j), lane owns cols (2*lane, 2*lane+1) ----
      float i0 = 0.f, i1 = 0.f;
#pragma unroll
      for (int e2 = 0; e2 < 16; ++e2) {
        float ge = (e2 < 8) ? gl[e2] : gh[e2 - 8];
        v2h pv = *(const v2h*)(Xw + e2 * XSTR + 64 + 2 * lane);
        i0 += ge * (float)pv[0];
        i1 += ge * (float)pv[1];
      }
      i0 *= (1.f / 16.f);
      i1 *= (1.f / 16.f);

      // ---- v_i @ W_w via WMMA with replicated-row A (B resident) ----
      v8f accw[4] = {};
#pragma unroll
      for (int c = 0; c < 2; ++c) {
        v16h a = frag_a(VIw, c, lane);  // all 16 rows = v_i
#pragma unroll
        for (int nt = 0; nt < 4; ++nt)
          accw[nt] = wmma16(a, bWW[c][nt], accw[nt]);
      }
      if (lane < 16) {
#pragma unroll
        for (int nt = 0; nt < 4; ++nt) VWw[nt * 16 + lane] = accw[nt][0];
      }
      LDS_FENCE();
      float w0 = VWw[2 * lane], w1v = VWw[2 * lane + 1];
      float u0 = i0 + w0;
      u0 = u0 > 0.f ? u0 : 0.f;
      float u1 = i1 + w1v;
      u1 = u1 > 0.f ? u1 : 0.f;
      // ---- LayerNorm over 64 values (2 per lane) ----
      float sm = u0 + u1;
#pragma unroll
      for (int m = 1; m < 32; m <<= 1) sm += __shfl_xor(sm, m, 32);
      float mean = sm * (1.f / 64.f);
      float d0 = u0 - mean, d1 = u1 - mean;
      float vq = d0 * d0 + d1 * d1;
#pragma unroll
      for (int m = 1; m < 32; m <<= 1) vq += __shfl_xor(vq, m, 32);
      float rstd = rsqrtf(vq * (1.f / 64.f) + 1e-5f);
      float2 o;
      o.x = d0 * rstd * lg0 + lb0v;
      o.y = d1 * rstd * lg1 + lb1v;
      *(float2*)(vout + i * 64 + 2 * lane) = o;
    }
    LDS_FENCE();
  }
}

extern "C" void kernel_launch(void* const* d_in, const int* in_sizes, int n_in,
                              void* d_out, int out_size, void* d_ws,
                              size_t ws_size, hipStream_t stream) {
  const float* coords = (const float*)d_in[0];
  const float* feat = (const float*)d_in[1];
  const int* idx = (const int*)d_in[2];
  const float* lift_w = (const float*)d_in[3];
  const float* lift_b = (const float*)d_in[4];
  const float* gw1 = (const float*)d_in[5];
  const float* gb1 = (const float*)d_in[6];
  const float* gw2 = (const float*)d_in[7];
  const float* gb2 = (const float*)d_in[8];
  const float* gw3 = (const float*)d_in[9];
  const float* gb3 = (const float*)d_in[10];
  const float* Ww = (const float*)d_in[11];
  const float* lng = (const float*)d_in[12];
  const float* lnb = (const float*)d_in[13];
  const float* pw1 = (const float*)d_in[14];
  const float* pb1 = (const float*)d_in[15];
  const float* pw2 = (const float*)d_in[16];
  const float* pb2 = (const float*)d_in[17];

  float* out = (float*)d_out;
  float* out_scores = out;                          // [N]
  float* out_idx = out + NPTS;                      // [N,K]
  float* out_wij = out + NPTS + NPTS * KNBR;        // [N,K]
  float* out_v = out + NPTS + 2 * NPTS * KNBR;      // [N,H]

  float* v0 = (float*)d_ws;
  float* v1 = v0 + (size_t)NPTS * HDIM;

  lift_kernel<<<(NPTS * HDIM + TPB - 1) / TPB, TPB, 0, stream>>>(
      coords, feat, lift_w, lift_b, v0);

  const int nw = GRID_STEP * WPB;
  // t=0: v0 -> v1
  sno_step<<<GRID_STEP, TPB, 0, stream>>>(v0, v1, coords, idx, gw1, gb1, gw2,
                                          gb2, gw3, gb3, Ww, lng, lnb, 0, pw1,
                                          pb1, pw2, pb2, out_scores, out_wij, 0,
                                          nw);
  // t=1: v1 -> v0
  sno_step<<<GRID_STEP, TPB, 0, stream>>>(v1, v0, coords, idx, gw1, gb1, gw2,
                                          gb2, gw3, gb3, Ww, lng, lnb, 1, pw1,
                                          pb1, pw2, pb2, out_scores, out_wij, 0,
                                          nw);
  // t=2: v0 -> v1
  sno_step<<<GRID_STEP, TPB, 0, stream>>>(v0, v1, coords, idx, gw1, gb1, gw2,
                                          gb2, gw3, gb3, Ww, lng, lnb, 2, pw1,
                                          pb1, pw2, pb2, out_scores, out_wij, 0,
                                          nw);
  // final pass: scores + w_ij from v1 (vout unused)
  sno_step<<<GRID_STEP, TPB, 0, stream>>>(v1, v0, coords, idx, gw1, gb1, gw2,
                                          gb2, gw3, gb3, Ww, lng, lnb, 0, pw1,
                                          pb1, pw2, pb2, out_scores, out_wij, 1,
                                          nw);
  tail_copy<<<(NPTS * HDIM + TPB - 1) / TPB, TPB, 0, stream>>>(idx, v1, out_idx,
                                                               out_v);
}
